// EnhancedCurvedMemory_42202348650969
// MI455X (gfx1250) — matrix operand order repaired and checked by reading, not compiled, not run on hardware
//
#include <hip/hip_runtime.h>
#include <hip/hip_bf16.h>
#include <math.h>

typedef __attribute__((ext_vector_type(16))) __bf16 bf16x16;
typedef __attribute__((ext_vector_type(8)))  float  f32x8;
typedef __attribute__((ext_vector_type(4)))  unsigned u32x4;
typedef __attribute__((ext_vector_type(8)))  unsigned u32x8;

#define TM 32
#define TN 256
#define TK 32
#define LDSS 40   // LDS row stride in ushort elements (80 bytes: 64B data + 16B TDM pad)

static __device__ __forceinline__ unsigned short f32_to_bf16(float f) {
    union { float f; unsigned u; } v; v.f = f;
    unsigned u = v.u;
    u += 0x7FFFu + ((u >> 16) & 1u);   // round-to-nearest-even
    return (unsigned short)(u >> 16);
}

// ---- fragment loaders (wave32, 16-bit WMMA layouts per CDNA5 ISA 7.12.2) ----
static __device__ __forceinline__ bf16x16 load_frag_a(const unsigned short* base, int lane) {
    int r = lane & 15, half = lane >> 4;
    const char* p = (const char*)(base + r * LDSS) + half * 16;
    union { uint4 u[2]; bf16x16 v; } f;
    f.u[0] = *(const uint4*)(p);        // K 0..7   (or 8..15)
    f.u[1] = *(const uint4*)(p + 32);   // K 16..23 (or 24..31)
    return f.v;
}
static __device__ __forceinline__ bf16x16 load_frag_b(const unsigned short* base, int lane) {
    int r = lane & 15, half = lane >> 4;
    const char* p = (const char*)(base + r * LDSS) + half * 16;
    union { uint4 u[2]; bf16x16 v; } f;
    f.u[0] = *(const uint4*)(p);        // K 0..7   (or 8..15)
    f.u[1] = *(const uint4*)(p + 32);   // K 16..23 (or 24..31)
    return f.v;
}

// ---- TDM: DMA one B tile (TN rows x 64B, row stride ldb*2 bytes) into LDS ----
// data_size=8B, tile_dim0=8 (64B), tile_dim1=TN; pad 16B after every 64B so the
// LDS image lands with an 80B row stride (bank-spread), matching LDSS.
static __device__ __forceinline__ void tdm_load_b_tile(const unsigned short* gptr,
                                                       int ldb, unsigned lds_byte_addr) {
    unsigned long long ga = (unsigned long long)(uintptr_t)gptr;
    u32x4 g0;
    g0[0] = 1u;                                                   // count=1 (valid)
    g0[1] = lds_byte_addr;                                        // lds_addr
    g0[2] = (unsigned)(ga & 0xffffffffull);                       // global_addr lo
    g0[3] = (unsigned)((ga >> 32) & 0x01ffffffull) | 0x80000000u; // addr hi | type=2
    const unsigned TD = 1u << 20;  // generous tensor dims (tiles are always in-bounds)
    u32x8 g1;
    g1[0] = (3u << 16)   // data_size = 8B
          | (1u << 20)   // pad_enable
          | (3u << 22)   // pad_interval: 16 DWORDs (64B)
          | (3u << 25);  // pad_amount : 4 DWORDs (16B)
    g1[1] = (TD & 0xffffu) << 16;                  // tensor_dim0[15:0]
    g1[2] = (TD >> 16) | ((TD & 0xffffu) << 16);   // tensor_dim0[31:16] | tensor_dim1[15:0]
    g1[3] = (TD >> 16) | (8u << 16);               // tensor_dim1[31:16] | tile_dim0=8
    g1[4] = (unsigned)TN;                          // tile_dim1=TN, tile_dim2=0
    g1[5] = (unsigned)(ldb >> 2);                  // tensor_dim0_stride = ldb*2/8 (8B units)
    g1[6] = 0u;
    g1[7] = 0u;
    asm volatile("tensor_load_to_lds %0, %1" :: "s"(g0), "s"(g1) : "memory");
}

static __device__ __forceinline__ void stage_a(unsigned short* As, int m, int kq, float4 v) {
    unsigned short pk[4] = { f32_to_bf16(v.x), f32_to_bf16(v.y),
                             f32_to_bf16(v.z), f32_to_bf16(v.w) };
    *(uint2*)(&As[m * LDSS + kq]) = *(const uint2*)pk;
}

static __device__ __forceinline__ void mma_tiles(const unsigned short* As,
                                                 const unsigned short* bs,
                                                 int lane, int wave, f32x8 c[2][2]) {
    bf16x16 a0 = load_frag_a(As,            lane);
    bf16x16 a1 = load_frag_a(As + 16*LDSS,  lane);
    bf16x16 b0 = load_frag_b(bs + (wave*32 +  0) * LDSS, lane);
    bf16x16 b1 = load_frag_b(bs + (wave*32 + 16) * LDSS, lane);
    c[0][0] = __builtin_amdgcn_wmma_f32_16x16x32_bf16(false, a0, false, b0, (short)0, c[0][0], false, false);
    c[0][1] = __builtin_amdgcn_wmma_f32_16x16x32_bf16(false, a0, false, b1, (short)0, c[0][1], false, false);
    c[1][0] = __builtin_amdgcn_wmma_f32_16x16x32_bf16(false, a1, false, b0, (short)0, c[1][0], false, false);
    c[1][1] = __builtin_amdgcn_wmma_f32_16x16x32_bf16(false, a1, false, b1, (short)0, c[1][1], false, false);
}

// ---------------- bf16 WMMA GEMM: C[M,N] = epi(A[M,K] x B[N,K]^T-layout) --------------
// EPI: 0 = store f32; 1 = tanh(x+bias); 2 = tanh(x+bias) then mean over TM rows -> 1 row
// Requires K % TK == 0 (all call sites: K = 1024 or 4096).
template <int EPI>
__global__ __launch_bounds__(256)
void gemm_bf16_wmma(const float* __restrict__ A, int lda,
                    const unsigned short* __restrict__ B, int ldb,  // bf16 [N][K]
                    const float* __restrict__ bias,
                    float* __restrict__ C, int ldc,
                    int K) {
    __shared__ __align__(16) unsigned short As[TM * LDSS];
    __shared__ __align__(16) unsigned short Bs[2][TN * LDSS];

    const int tid  = threadIdx.x;
    const int lane = tid & 31;
    const int wave = tid >> 5;                 // 0..7, owns 32 columns
    const int row0 = blockIdx.y * TM;
    const int col0 = blockIdx.x * TN;

    const int am  = tid >> 3;                  // A-stage row (0..31)
    const int akq = (tid & 7) << 2;            // A-stage K offset (0,4,..,28)
    const float* arow = A + (size_t)(row0 + am) * lda + akq;

    f32x8 c[2][2] = {};

    // prime the pipelines: B tile k0=0 via TDM, A tile k0=0 in registers
    if (wave == 0)
        tdm_load_b_tile(B + (size_t)col0 * ldb, ldb, (unsigned)(uintptr_t)(&Bs[0][0]));
    float4 av = *(const float4*)(arow);

    #pragma unroll 1
    for (int k0 = 0; k0 < K; k0 += 2 * TK) {
        // ---------- phase 0: compute on Bs[0], prefetch k0+TK -> Bs[1] ----------
        __syncthreads();                       // As free, Bs[1] free
        stage_a(As, am, akq, av);
        const bool more1 = (k0 + TK) < K;
        if (more1) av = *(const float4*)(arow + k0 + TK);          // hide load latency
        if (wave == 0) {
            if (more1) {
                tdm_load_b_tile(B + (size_t)col0 * ldb + (k0 + TK), ldb,
                                (unsigned)(uintptr_t)(&Bs[1][0]));
                __builtin_amdgcn_s_wait_tensorcnt(1);  // oldest (Bs[0]) retired
            } else {
                __builtin_amdgcn_s_wait_tensorcnt(0);
            }
        }
        __syncthreads();
        mma_tiles(As, &Bs[0][0], lane, wave, c);
        if (!more1) break;

        // ---------- phase 1: compute on Bs[1], prefetch k0+2*TK -> Bs[0] ----------
        __syncthreads();                       // As free, Bs[0] free
        stage_a(As, am, akq, av);
        const bool more2 = (k0 + 2 * TK) < K;
        if (more2) av = *(const float4*)(arow + k0 + 2 * TK);
        if (wave == 0) {
            if (more2) {
                tdm_load_b_tile(B + (size_t)col0 * ldb + (k0 + 2 * TK), ldb,
                                (unsigned)(uintptr_t)(&Bs[0][0]));
                __builtin_amdgcn_s_wait_tensorcnt(1);  // oldest (Bs[1]) retired
            } else {
                __builtin_amdgcn_s_wait_tensorcnt(0);
            }
        }
        __syncthreads();
        mma_tiles(As, &Bs[1][0], lane, wave, c);
    }

    // C layout: vgpr i, lanes 0-15: M=i, N=lane ; lanes 16-31: M=8+i, N=lane-16
    const int nl = lane & 15, half = lane >> 4;
    if (EPI == 2) {
        // tanh + mean over the 32 rows (one sequence group) -> query row (row0/TM)
        #pragma unroll
        for (int ni = 0; ni < 2; ++ni) {
            const int col = col0 + wave * 32 + ni * 16 + nl;
            const float bb = bias[col];
            float s = 0.f;
            #pragma unroll
            for (int mi = 0; mi < 2; ++mi)
                #pragma unroll
                for (int i = 0; i < 8; ++i)
                    s += tanhf(c[mi][ni][i] + bb);
            s += __shfl_xor(s, 16, 32);        // combine both M-halves (same column)
            if (half == 0)
                C[(size_t)(row0 / TM) * ldc + col] = s * (1.0f / TM);
        }
    } else {
        #pragma unroll
        for (int ni = 0; ni < 2; ++ni) {
            const int col = col0 + wave * 32 + ni * 16 + nl;
            const float bb = (EPI == 1) ? bias[col] : 0.f;
            #pragma unroll
            for (int mi = 0; mi < 2; ++mi) {
                const int rbase = row0 + mi * 16 + half * 8;
                #pragma unroll
                for (int i = 0; i < 8; ++i) {
                    float v = c[mi][ni][i] + bb;
                    if (EPI == 1) v = tanhf(v);
                    C[(size_t)(rbase + i) * ldc + col] = v;
                }
            }
        }
    }
}

// ---------------- prep: f32 [R,C] -> bf16 [C,R] (transpose + convert) ----------------
__global__ __launch_bounds__(256)
void transpose_to_bf16(const float* __restrict__ in, unsigned short* __restrict__ out,
                       int R, int C) {
    __shared__ float tile[32][33];
    const int c0 = blockIdx.x * 32, r0 = blockIdx.y * 32;
    const int tx = threadIdx.x & 31, ty = threadIdx.x >> 5;
    for (int i = ty; i < 32; i += 8)
        tile[i][tx] = in[(size_t)(r0 + i) * C + c0 + tx];
    __syncthreads();
    for (int i = ty; i < 32; i += 8)
        out[(size_t)(c0 + i) * R + r0 + tx] = f32_to_bf16(tile[tx][i]);
}

// ---------------- prep: f32 -> bf16 same layout ----------------
__global__ __launch_bounds__(256)
void convert_to_bf16(const float* __restrict__ in, unsigned short* __restrict__ out, size_t n4) {
    size_t i = ((size_t)blockIdx.x * 256 + threadIdx.x);
    if (i >= n4) return;
    const float4 v = *(const float4*)(in + i * 4);
    unsigned short pk[4] = { f32_to_bf16(v.x), f32_to_bf16(v.y),
                             f32_to_bf16(v.z), f32_to_bf16(v.w) };
    *(uint2*)(out + i * 4) = *(const uint2*)pk;
}

// ---------------- gate: scale[b] = (0.5 + sigmoid(q.w_curv + b_curv)) / max(T,1e-6) ----
__global__ __launch_bounds__(256)
void gate_scale_kernel(const float* __restrict__ query, const float* __restrict__ w_curv,
                       const float* __restrict__ b_curv, const float* __restrict__ temp,
                       float* __restrict__ scale, int H) {
    __shared__ float red[256];
    const int b = blockIdx.x;
    float s = 0.f;
    for (int h = threadIdx.x; h < H; h += 256) s += query[(size_t)b * H + h] * w_curv[h];
    red[threadIdx.x] = s; __syncthreads();
    for (int st = 128; st > 0; st >>= 1) {
        if (threadIdx.x < st) red[threadIdx.x] += red[threadIdx.x + st];
        __syncthreads();
    }
    if (threadIdx.x == 0) {
        float g = 1.f / (1.f + __expf(-(red[0] + b_curv[0])));
        scale[b] = (0.5f + g) / fmaxf(temp[0], 1e-6f);
    }
}

// ---------------- top-32 of 4096 (iterative argmax in LDS), sorted desc ----------------
__global__ __launch_bounds__(256)
void topk_kernel(const float* __restrict__ act, const float* __restrict__ scale,
                 float* __restrict__ vals, int* __restrict__ idx) {
    __shared__ float sv[4096];
    __shared__ float rv[256];
    __shared__ int   ri[256];
    const int b = blockIdx.x;
    const float sc = scale[b];
    for (int m = threadIdx.x; m < 4096; m += 256)
        sv[m] = act[(size_t)b * 4096 + m] * sc;
    __syncthreads();
    for (int k = 0; k < 32; ++k) {
        float best = -INFINITY; int bi = 0x7FFFFFFF;
        for (int m = threadIdx.x; m < 4096; m += 256) {
            float v = sv[m];
            if (v > best) { best = v; bi = m; }
        }
        rv[threadIdx.x] = best; ri[threadIdx.x] = bi;
        __syncthreads();
        for (int st = 128; st > 0; st >>= 1) {
            if (threadIdx.x < st) {
                float ov = rv[threadIdx.x + st]; int oi = ri[threadIdx.x + st];
                if (ov > rv[threadIdx.x] || (ov == rv[threadIdx.x] && oi < ri[threadIdx.x])) {
                    rv[threadIdx.x] = ov; ri[threadIdx.x] = oi;
                }
            }
            __syncthreads();
        }
        if (threadIdx.x == 0) {
            vals[(size_t)b * 32 + k] = rv[0];
            idx [(size_t)b * 32 + k] = ri[0];
            sv[ri[0]] = -INFINITY;
        }
        __syncthreads();
    }
}

// ---------------- row softmax over 4096 ----------------
__global__ __launch_bounds__(256)
void softmax_kernel(const float* __restrict__ in, float* __restrict__ out) {
    __shared__ float sv[4096];
    __shared__ float red[256];
    const int b = blockIdx.x;
    float mx = -INFINITY;
    for (int m = threadIdx.x; m < 4096; m += 256) {
        float v = in[(size_t)b * 4096 + m]; sv[m] = v; mx = fmaxf(mx, v);
    }
    red[threadIdx.x] = mx; __syncthreads();
    for (int st = 128; st > 0; st >>= 1) {
        if (threadIdx.x < st) red[threadIdx.x] = fmaxf(red[threadIdx.x], red[threadIdx.x + st]);
        __syncthreads();
    }
    mx = red[0]; __syncthreads();
    float sum = 0.f;
    for (int m = threadIdx.x; m < 4096; m += 256) {
        float e = __expf(sv[m] - mx); sv[m] = e; sum += e;
    }
    red[threadIdx.x] = sum; __syncthreads();
    for (int st = 128; st > 0; st >>= 1) {
        if (threadIdx.x < st) red[threadIdx.x] += red[threadIdx.x + st];
        __syncthreads();
    }
    const float inv = 1.f / red[0];
    for (int m = threadIdx.x; m < 4096; m += 256)
        out[(size_t)b * 4096 + m] = sv[m] * inv;
}

// ------- comb = softmax(vals + act[idx]) over K=32; read = comb . mem[idx] -------------
__global__ __launch_bounds__(256)
void combine_read_kernel(const float* __restrict__ vals, const int* __restrict__ idx,
                         const float* __restrict__ act, const float* __restrict__ mem,
                         float* __restrict__ readout, int H) {
    __shared__ float comb[32];
    __shared__ int   sidx[32];
    const int b = blockIdx.x, t = threadIdx.x;
    if (t < 32) {
        int id = idx[(size_t)b * 32 + t];
        sidx[t] = id;
        comb[t] = vals[(size_t)b * 32 + t] + act[(size_t)b * 4096 + id];
    }
    __syncthreads();
    if (t == 0) {
        float mx = comb[0];
        for (int k = 1; k < 32; ++k) mx = fmaxf(mx, comb[k]);
        float s = 0.f;
        for (int k = 0; k < 32; ++k) { comb[k] = __expf(comb[k] - mx); s += comb[k]; }
        float inv = 1.f / s;
        for (int k = 0; k < 32; ++k) comb[k] *= inv;
    }
    __syncthreads();
    for (int h = t; h < H; h += 256) {
        float s = 0.f;
        #pragma unroll
        for (int k = 0; k < 32; ++k)
            s += comb[k] * mem[(size_t)sidx[k] * H + h];
        readout[(size_t)b * H + h] = s;
    }
}

// ---------------- broadcast [B,D] -> [B,S,D] ----------------
__global__ __launch_bounds__(256)
void broadcast_kernel(const float* __restrict__ row, float* __restrict__ out,
                      int S, int D4) {
    size_t i = (size_t)blockIdx.x * 256 + threadIdx.x;   // float4 index into out
    int d4 = (int)(i % D4);
    int b  = (int)((i / D4) / S);
    const float4 v = *(const float4*)(row + (size_t)b * (D4 * 4) + d4 * 4);
    *(float4*)(out + i * 4) = v;
}

extern "C" void kernel_launch(void* const* d_in, const int* in_sizes, int n_in,
                              void* d_out, int out_size, void* d_ws, size_t ws_size,
                              hipStream_t stream) {
    constexpr int B = 2048, S = 32, D = 1024, H = 1024, M = 4096;

    const float* x      = (const float*)d_in[0];
    const float* W_enc  = (const float*)d_in[2];
    const float* b_enc  = (const float*)d_in[3];
    const float* w_curv = (const float*)d_in[4];
    const float* b_curv = (const float*)d_in[5];
    const float* mem    = (const float*)d_in[6];
    const float* assoc  = (const float*)d_in[7];
    const float* W_dec  = (const float*)d_in[8];
    const float* b_dec  = (const float*)d_in[9];
    const float* temp   = (const float*)d_in[10];
    float* out = (float*)d_out;

    char* w = (char*)d_ws;
    auto alloc = [&](size_t bytes) -> char* {
        char* p = w; w += (bytes + 255) & ~(size_t)255; return p;
    };
    unsigned short* WencT  = (unsigned short*)alloc((size_t)D * H * 2);  // [H][D]
    unsigned short* memb   = (unsigned short*)alloc((size_t)M * H * 2);  // [M][H]
    unsigned short* assocT = (unsigned short*)alloc((size_t)M * M * 2);  // [M][M] transposed
    unsigned short* WdecT  = (unsigned short*)alloc((size_t)H * D * 2);  // [D][H]
    float* query  = (float*)alloc((size_t)B * H * 4);
    float* scale  = (float*)alloc((size_t)B * 4);
    float* act_a  = (float*)alloc((size_t)B * M * 4);
    float* act_b  = (float*)alloc((size_t)B * M * 4);
    float* vals   = (float*)alloc((size_t)B * 32 * 4);
    int*   idx    = (int*)  alloc((size_t)B * 32 * 4);
    float* readb  = (float*)alloc((size_t)B * H * 4);
    float* outrow = (float*)alloc((size_t)B * D * 4);

    // ---- prep: convert/transpose weights to bf16 ----
    transpose_to_bf16<<<dim3(H/32, D/32), 256, 0, stream>>>(W_enc, WencT, D, H);
    convert_to_bf16  <<<dim3((M*H/4 + 255)/256), 256, 0, stream>>>(mem, memb, (size_t)M*H/4);
    transpose_to_bf16<<<dim3(M/32, M/32), 256, 0, stream>>>(assoc, assocT, M, M);
    transpose_to_bf16<<<dim3(D/32, H/32), 256, 0, stream>>>(W_dec, WdecT, H, D);

    // ---- 1) enc = tanh(x @ W_enc + b_enc); query = mean over S (fused) ----
    gemm_bf16_wmma<2><<<dim3(H/TN, (B*S)/TM), 256, 0, stream>>>(
        x, D, WencT, D, b_enc, query, H, D);

    // ---- gate scale ----
    gate_scale_kernel<<<dim3(B), 256, 0, stream>>>(query, w_curv, b_curv, temp, scale, H);

    // ---- 2) act_a = query @ mem^T ----
    gemm_bf16_wmma<0><<<dim3(M/TN, B/TM), 256, 0, stream>>>(
        query, H, memb, H, nullptr, act_a, M, H);

    // ---- top-k on scaled sim ----
    topk_kernel<<<dim3(B), 256, 0, stream>>>(act_a, scale, vals, idx);

    // ---- two rounds: softmax -> @ assoc ----
    softmax_kernel<<<dim3(B), 256, 0, stream>>>(act_a, act_b);
    gemm_bf16_wmma<0><<<dim3(M/TN, B/TM), 256, 0, stream>>>(
        act_b, M, assocT, M, nullptr, act_a, M, M);
    softmax_kernel<<<dim3(B), 256, 0, stream>>>(act_a, act_b);
    gemm_bf16_wmma<0><<<dim3(M/TN, B/TM), 256, 0, stream>>>(
        act_b, M, assocT, M, nullptr, act_a, M, M);

    // ---- combine + weighted gather read ----
    combine_read_kernel<<<dim3(B), 256, 0, stream>>>(vals, idx, act_a, mem, readb, H);

    // ---- 4) outrow = tanh(read @ W_dec + b_dec) ----
    gemm_bf16_wmma<1><<<dim3(D/TN, B/TM), 256, 0, stream>>>(
        readb, H, WdecT, H, b_dec, outrow, D, H);

    // ---- broadcast to [B,S,D] ----
    broadcast_kernel<<<dim3((unsigned)((size_t)B * S * (D/4) / 256)), 256, 0, stream>>>(
        outrow, out, S, D/4);
}
